// Attention_60129542144524
// MI455X (gfx1250) — compile-verified, matrix operand order checked by reading, not run on hardware
//
#include <hip/hip_runtime.h>

// ---------------------------------------------------------------------------
// Types for CDNA5 WMMA (wave32): 16x16x32 bf16 -> f32 accumulate
// ---------------------------------------------------------------------------
typedef __bf16 bf16_t;
typedef __attribute__((ext_vector_type(16))) __bf16 bf16x16;
typedef __attribute__((ext_vector_type(8)))  __bf16 bf16x8;
typedef __attribute__((ext_vector_type(8)))  float  f32x8;

// The async-to-LDS builtin takes pointers to 16-byte int vectors (observed
// from the clang diagnostic: 'int __vector(4) *').
typedef int v4i_vs __attribute__((vector_size(16)));

union FragU { bf16x16 v; bf16x8 h[2]; };

// ---- async global->LDS path (guarded: falls back to sync copy) ------------
#if defined(__has_builtin)
#if __has_builtin(__builtin_amdgcn_global_load_async_to_lds_b128) && \
    __has_builtin(__builtin_amdgcn_s_wait_asynccnt)
#define USE_ASYNC_LDS 1
#endif
#endif
#ifndef USE_ASYNC_LDS
#define USE_ASYNC_LDS 0
#endif

__device__ __forceinline__ void copy16_g2l(const bf16_t* g, bf16_t* l) {
#if USE_ASYNC_LDS
    __builtin_amdgcn_global_load_async_to_lds_b128(
        (v4i_vs*)const_cast<bf16_t*>(g), (v4i_vs*)l, 0, 0);
#else
    *(bf16x8*)l = *(const bf16x8*)g;
#endif
}
__device__ __forceinline__ void wait_async1() {
#if USE_ASYNC_LDS
    __builtin_amdgcn_s_wait_asynccnt(1);
#endif
}
__device__ __forceinline__ void wait_async0() {
#if USE_ASYNC_LDS
    __builtin_amdgcn_s_wait_asynccnt(0);
#endif
}

__device__ __forceinline__ f32x8 zero8() {
    f32x8 z;
#pragma unroll
    for (int i = 0; i < 8; ++i) z[i] = 0.0f;
    return z;
}

__device__ __forceinline__ f32x8 wmma_bf16(bf16x16 a, bf16x16 b, f32x8 c) {
    return __builtin_amdgcn_wmma_f32_16x16x32_bf16(
        false, a, false, b, (short)0, c, false, false);
}

// 16-bit A/B fragment (16x32): lane L holds row L%16;
// lanes 0-15 take K[0..7]+K[16..23], lanes 16-31 take K[8..15]+K[24..31].
// off = row*ld + kbase + ((lane>>4)<<3), 32-bit vs a uniform base (GVS form).
__device__ __forceinline__ bf16x16 load_frag_u32(const bf16_t* __restrict__ base,
                                                 unsigned off) {
    FragU f;
    f.h[0] = *(const bf16x8*)(base + off);
    f.h[1] = *(const bf16x8*)(base + off + 16);
    return f.v;
}

// ---------------------------------------------------------------------------
// fp32 -> bf16 convert / transpose-convert
// ---------------------------------------------------------------------------
__global__ __launch_bounds__(256) void cvt_f32_bf16(const float* __restrict__ src,
                                                    bf16_t* __restrict__ dst, int n) {
    int i = blockIdx.x * 256 + threadIdx.x;
    if (i < n) dst[i] = (bf16_t)src[i];
}

// dst[o*K + i] = src[i*src_ld + col_off + o]
__global__ __launch_bounds__(256) void cvt_transpose(const float* __restrict__ src,
                                                     bf16_t* __restrict__ dst,
                                                     int K, int src_ld, int col_off, int Nout) {
    int idx = blockIdx.x * 256 + threadIdx.x;
    if (idx >= Nout * K) return;
    int o = idx / K;
    int i = idx - o * K;
    dst[(size_t)o * K + i] = (bf16_t)src[(size_t)i * src_ld + col_off + o];
}

// ---------------------------------------------------------------------------
// WMMA GEMM: C[M x 1024] = A[M x K] * B, B stored as B^T row-major [1024 x K].
// Block = 256 thr (8 waves); block tile 256(M) x 64(N); wave tile 32 x 64
// (8 accumulator frags, 8 WMMAs per 32-wide k-step).
// B tile (64x32) staged in LDS via async global->LDS copy, double buffered,
// padded row stride 40 elems to spread LDS banks.
// Store modes: 0/1 bf16 [b,h,n,d] (q/k) | 2 bf16 [b,h,d,n] (v^T)
//              3 fp32 + bias, row-major (final projection)
// ---------------------------------------------------------------------------
#define BPAD 40

__global__ __launch_bounds__(256)
void gemm_bf16_wmma(const bf16_t* __restrict__ A, const bf16_t* __restrict__ B,
                    void* __restrict__ OUT, const float* __restrict__ bias,
                    int mode) {
    constexpr int K = 1024;
    __shared__ __align__(16) bf16_t bsh[2][64 * BPAD];   // 10 KB

    const int tid  = threadIdx.x;
    const int lane = tid & 31;
    const int wave = tid >> 5;
    const int n0   = blockIdx.x * 64;
    const int m0   = blockIdx.y * 256 + wave * 32;
    const int frow = lane & 15;
    const int khalf = (lane >> 4) << 3;

    // cooperative B-tile copy: thread t moves 16B; row = t/4, seg = (t%4)*8
    const int crow = tid >> 2;
    const int cseg = (tid & 3) * 8;
    const bf16_t* bsrc = B + (unsigned)((n0 + crow) * K + cseg);
    bf16_t* bdst0 = &bsh[0][crow * BPAD + cseg];
    bf16_t* bdst1 = &bsh[1][crow * BPAD + cseg];

    f32x8 c[2][4];
#pragma unroll
    for (int mi = 0; mi < 2; ++mi)
#pragma unroll
        for (int t = 0; t < 4; ++t) c[mi][t] = zero8();

    const unsigned aoff0 = (unsigned)((m0 + frow) * K + khalf);
    const unsigned aoff1 = aoff0 + 16u * K;

    copy16_g2l(bsrc, bdst0);
    int cur = 0;
    for (int k0 = 0; k0 < K; k0 += 32) {
        if (k0 + 32 < K) {                       // prefetch next tile, keep 1 in flight
            copy16_g2l(bsrc + (unsigned)(k0 + 32), cur ? bdst0 : bdst1);
            wait_async1();
        } else {
            wait_async0();
        }
        __syncthreads();                          // tile `cur` visible to all waves

        FragU a0, a1;
        a0.h[0] = *(const bf16x8*)(A + (aoff0 + k0));
        a0.h[1] = *(const bf16x8*)(A + (aoff0 + k0 + 16));
        a1.h[0] = *(const bf16x8*)(A + (aoff1 + k0));
        a1.h[1] = *(const bf16x8*)(A + (aoff1 + k0 + 16));

        const bf16_t* bb = &bsh[cur][0];
#pragma unroll
        for (int t = 0; t < 4; ++t) {
            FragU bfm;
            const bf16_t* bp = bb + ((t * 16 + frow) * BPAD + khalf);
            bfm.h[0] = *(const bf16x8*)(bp);
            bfm.h[1] = *(const bf16x8*)(bp + 16);
            c[0][t] = wmma_bf16(a0.v, bfm.v, c[0][t]);
            c[1][t] = wmma_bf16(a1.v, bfm.v, c[1][t]);
        }
        __syncthreads();                          // done reading before overwrite
        cur ^= 1;
    }

    // C layout: VGPR r, lane L -> (m_local = r + 8*(L/16), n_local = L%16)
    const int mrow_off = (lane >> 4) << 3;
    if (mode == 3) {
        float* out = (float*)OUT;
#pragma unroll
        for (int mi = 0; mi < 2; ++mi)
#pragma unroll
            for (int t = 0; t < 4; ++t) {
                int n = n0 + t * 16 + frow;
                float bv = bias[n];
#pragma unroll
                for (int r = 0; r < 8; ++r) {
                    int m = m0 + mi * 16 + r + mrow_off;
                    out[(size_t)m * 1024 + n] = c[mi][t][r] + bv;
                }
            }
    } else {
        bf16_t* out = (bf16_t*)OUT;
#pragma unroll
        for (int mi = 0; mi < 2; ++mi)
#pragma unroll
            for (int t = 0; t < 4; ++t) {
                int n = n0 + t * 16 + frow;
                int head = n >> 6, d = n & 63;
#pragma unroll
                for (int r = 0; r < 8; ++r) {
                    int m = m0 + mi * 16 + r + mrow_off;
                    int bidx = m >> 11, tok = m & 2047;
                    size_t idx;
                    if (mode == 2)
                        idx = (((size_t)bidx * 16 + head) * 64 + d) * 2048 + tok; // [b,h,d,n]
                    else
                        idx = (((size_t)bidx * 16 + head) * 2048 + tok) * 64 + d; // [b,h,n,d]
                    out[idx] = (bf16_t)c[mi][t][r];
                }
            }
    }
}

// ---------------------------------------------------------------------------
// Flash attention, online softmax. Grid (n/128, b*h); 8 waves/block, each
// wave owns 16 query rows. 32-key chunks: 4 WMMAs for S, shfl_xor row
// reductions in each 16-lane half, P reshaped C->A layout via padded
// wave-private LDS, 4 WMMAs for P*V.
// ---------------------------------------------------------------------------
#define SEQ 2048
#define PPAD 40

__global__ __launch_bounds__(256)
void flash_attn_wmma(const bf16_t* __restrict__ Q, const bf16_t* __restrict__ Kv,
                     const bf16_t* __restrict__ VT, bf16_t* __restrict__ AO) {
    __shared__ __align__(16) bf16_t plds[8][16 * PPAD];   // 10 KB

    const int lane = threadIdx.x & 31;
    const int wave = threadIdx.x >> 5;
    const int bh = blockIdx.y;
    const int b = bh >> 4, h = bh & 15;
    const int m0 = blockIdx.x * 128 + wave * 16;
    const int frow = lane & 15;
    const int khalf = (lane >> 4) << 3;
    const int rlo = khalf;                      // 0 or 8: row offset of lane group

    const bf16_t* q  = Q  + (size_t)bh * SEQ * 64;
    const bf16_t* k  = Kv + (size_t)bh * SEQ * 64;
    const bf16_t* vt = VT + (size_t)bh * 64 * SEQ;      // [d, n]

    const unsigned qoff = (unsigned)((m0 + frow) * 64 + khalf);
    const bf16x16 qa0 = load_frag_u32(q, qoff);
    const bf16x16 qa1 = load_frag_u32(q, qoff + 32);

    f32x8 o0 = zero8(), o1 = zero8(), o2 = zero8(), o3 = zero8();
    float mrow[8], lrow[8];
#pragma unroll
    for (int r = 0; r < 8; ++r) { mrow[r] = -3.0e38f; lrow[r] = 0.0f; }

    const float scale = 0.125f;                 // 1/sqrt(64)
    const unsigned kfix = (unsigned)(frow * 64 + khalf);
    const unsigned vfix = (unsigned)(frow * SEQ + khalf);

    for (int j0 = 0; j0 < SEQ; j0 += 32) {
        // ---- S = Q*K^T (16 x 32), d=64 in two k-steps ----
        f32x8 s0 = zero8(), s1 = zero8();
        {
            const unsigned kb0 = (unsigned)(j0 * 64) + kfix;
            s0 = wmma_bf16(qa0, load_frag_u32(k, kb0),             s0);
            s0 = wmma_bf16(qa1, load_frag_u32(k, kb0 + 32),        s0);
            s1 = wmma_bf16(qa0, load_frag_u32(k, kb0 + 1024),      s1);   // +16 rows
            s1 = wmma_bf16(qa1, load_frag_u32(k, kb0 + 1024 + 32), s1);
        }
        // ---- online softmax update over 32 keys ----
#pragma unroll
        for (int r = 0; r < 8; ++r) {
            float a0 = s0[r] * scale;
            float a1 = s1[r] * scale;
            float cm = fmaxf(a0, a1);
            cm = fmaxf(cm, __shfl_xor(cm, 1));
            cm = fmaxf(cm, __shfl_xor(cm, 2));
            cm = fmaxf(cm, __shfl_xor(cm, 4));
            cm = fmaxf(cm, __shfl_xor(cm, 8));          // row max in 16-lane half
            float mnew  = fmaxf(mrow[r], cm);
            float alpha = __expf(mrow[r] - mnew);
            float p0 = __expf(a0 - mnew);
            float p1 = __expf(a1 - mnew);
            float rs = p0 + p1;
            rs += __shfl_xor(rs, 1);
            rs += __shfl_xor(rs, 2);
            rs += __shfl_xor(rs, 4);
            rs += __shfl_xor(rs, 8);                    // row sum in 16-lane half
            lrow[r] = lrow[r] * alpha + rs;
            mrow[r] = mnew;
            o0[r] *= alpha; o1[r] *= alpha; o2[r] *= alpha; o3[r] *= alpha;
            int row = r + rlo;
            plds[wave][row * PPAD + frow]      = (bf16_t)p0;
            plds[wave][row * PPAD + 16 + frow] = (bf16_t)p1;
        }
        // ---- reload P as A-fragment (wave-private LDS, in-order DS pipe) ----
        FragU pf;
        {
            const bf16_t* pp = &plds[wave][frow * PPAD + khalf];
            pf.h[0] = *(const bf16x8*)(pp);
            pf.h[1] = *(const bf16x8*)(pp + 16);
        }
        // ---- O += P * V via V^T[d, n] ----
        const unsigned vb = (unsigned)j0 + vfix;
        o0 = wmma_bf16(pf.v, load_frag_u32(vt, vb),                o0);
        o1 = wmma_bf16(pf.v, load_frag_u32(vt, vb + 16u * SEQ),    o1);
        o2 = wmma_bf16(pf.v, load_frag_u32(vt, vb + 32u * SEQ),    o2);
        o3 = wmma_bf16(pf.v, load_frag_u32(vt, vb + 48u * SEQ),    o3);
    }

    // ---- normalize + store bf16 to [b, n, h*64+d] ----
#pragma unroll
    for (int r = 0; r < 8; ++r) {
        float inv = 1.0f / lrow[r];
        int tok = m0 + r + rlo;
        size_t rowbase = ((size_t)b * SEQ + tok) * 1024 + h * 64;
        AO[rowbase +      frow] = (bf16_t)(o0[r] * inv);
        AO[rowbase + 16 + frow] = (bf16_t)(o1[r] * inv);
        AO[rowbase + 32 + frow] = (bf16_t)(o2[r] * inv);
        AO[rowbase + 48 + frow] = (bf16_t)(o3[r] * inv);
    }
}

// ---------------------------------------------------------------------------
// Host-side orchestration
// ---------------------------------------------------------------------------
extern "C" void kernel_launch(void* const* d_in, const int* in_sizes, int n_in,
                              void* d_out, int out_size, void* d_ws, size_t ws_size,
                              hipStream_t stream) {
    const float* x   = (const float*)d_in[0];   // [4,2048,1024]
    const float* Wq  = (const float*)d_in[1];   // [1024,1024]
    const float* Wkv = (const float*)d_in[2];   // [1024,2048]
    const float* Wo  = (const float*)d_in[3];   // [1024,1024]
    const float* bo  = (const float*)d_in[4];   // [1024]

    const size_t MT = 8192;                     // 4*2048 tokens
    const size_t DM = 1024;

    bf16_t* ws  = (bf16_t*)d_ws;                // ~88 MB bf16 staging
    bf16_t* xb  = ws;                           // [8192,1024]
    bf16_t* WqT = xb  + MT * DM;                // [1024,1024] (out,in)
    bf16_t* WkT = WqT + DM * DM;
    bf16_t* WvT = WkT + DM * DM;
    bf16_t* WoT = WvT + DM * DM;
    bf16_t* qb  = WoT + DM * DM;                // [b,h,n,d]
    bf16_t* kb  = qb  + MT * DM;                // [b,h,n,d]
    bf16_t* vtb = kb  + MT * DM;                // [b,h,d,n]
    bf16_t* aob = vtb + MT * DM;                // [b,n,h*d]

    cvt_f32_bf16<<<(int)(MT * DM / 256), 256, 0, stream>>>(x, xb, (int)(MT * DM));
    cvt_transpose<<<(int)(DM * DM / 256), 256, 0, stream>>>(Wq,  WqT, 1024, 1024, 0,    1024);
    cvt_transpose<<<(int)(DM * DM / 256), 256, 0, stream>>>(Wkv, WkT, 1024, 2048, 0,    1024);
    cvt_transpose<<<(int)(DM * DM / 256), 256, 0, stream>>>(Wkv, WvT, 1024, 2048, 1024, 1024);
    cvt_transpose<<<(int)(DM * DM / 256), 256, 0, stream>>>(Wo,  WoT, 1024, 1024, 0,    1024);

    dim3 g(16, 32), blk(256);
    gemm_bf16_wmma<<<g, blk, 0, stream>>>(xb, WqT, qb,  nullptr, 0);   // q
    gemm_bf16_wmma<<<g, blk, 0, stream>>>(xb, WkT, kb,  nullptr, 1);   // k
    gemm_bf16_wmma<<<g, blk, 0, stream>>>(xb, WvT, vtb, nullptr, 2);   // v^T

    flash_attn_wmma<<<dim3(16, 64), blk, 0, stream>>>(qb, kb, vtb, aob);

    gemm_bf16_wmma<<<g, blk, 0, stream>>>(aob, WoT, d_out, bo, 3);     // out proj + bias
}
